// SmilesGenerator_63711544869443
// MI455X (gfx1250) — matrix operand order repaired and checked by reading, not compiled, not run on hardware
//
#include <hip/hip_runtime.h>
#include <hip/hip_bf16.h>

// ---------------------------------------------------------------------------
// Model constants (fixed by the reference)
// ---------------------------------------------------------------------------
constexpr int E    = 1024;
constexpr int NH   = 8;
constexpr int DH   = 128;      // E / NH
constexpr int SEQ  = 256;
constexpr int BSZ  = 32;
constexpr int FF   = 4096;
constexpr int NL   = 6;
constexpr int VOC  = 45;
constexpr int MTOK = BSZ * SEQ;          // 8192 token rows
constexpr int MAXD = 250;                // MAX_LEN clamp for distance

typedef __attribute__((ext_vector_type(16))) __bf16 v16bf;
typedef __attribute__((ext_vector_type(8)))  float  v8f;

// ---------------------------------------------------------------------------
// GEMM: Y[m,n] = sum_k A_bf16[m,k] * W_bf16[n,k]  (+ bias / epilogue)
// Block tile 128x128, K-step 32 (= one bf16 WMMA). 256 threads = 8 waves,
// wave grid 2(M) x 4(N), each wave computes 4x2 tiles of 16x16.
// Fast path (GUARD=false): async global->LDS loads, double-buffered LDS,
// software-pipelined with the drain iteration peeled out of the hot loop.
// ---------------------------------------------------------------------------
constexpr int BM  = 128;
constexpr int BN  = 128;
constexpr int BK  = 32;
constexpr int BKP = 40;    // LDS row stride in elements (80B, 16B-aligned)

enum { EPI_F32 = 0, EPI_GELU = 1, EPI_QKV = 2, EPI_SCORES = 3, EPI_ATTNOUT = 4 };

union Frag {
    v16bf v;
    uint4 u[2];
};

template <int EPI, bool GUARD>
__global__ __launch_bounds__(256) void gemm_kernel(
    const __bf16* __restrict__ A,    // [batch][M,K]
    const __bf16* __restrict__ W,    // [batch][N,K]   (row n = output neuron n)
    const float*  __restrict__ bias, // [N] or null
    float*        __restrict__ outF,
    __bf16*       __restrict__ outB,
    __bf16*       __restrict__ outB2,
    __bf16*       __restrict__ outB3,
    int M, int N, int K,
    long asb, long wsb,              // batch strides (elements)
    float scale,
    const float* __restrict__ distEmb)
{
    __shared__ __align__(16) __bf16 As[2][BM * BKP];
    __shared__ __align__(16) __bf16 Ws[2][BN * BKP];

    const int tid  = threadIdx.x;
    const int bm0  = blockIdx.y * BM;
    const int bn0  = blockIdx.x * BN;
    const __bf16* Ab = A + (long)blockIdx.z * asb;
    const __bf16* Wb = W + (long)blockIdx.z * wsb;

    const int wave = tid >> 5;
    const int lane = tid & 31;
    const int wm   = wave >> 2;      // 0..1  -> 64 rows each
    const int wn   = wave & 3;       // 0..3  -> 32 cols each
    const int half = lane >> 4;      // lane group (0: lanes 0-15, 1: lanes 16-31)
    const int lrow = lane & 15;

    v8f zero = {};
    v8f acc[4][2];
#pragma unroll
    for (int i = 0; i < 4; ++i)
#pragma unroll
        for (int j = 0; j < 2; ++j) acc[i][j] = zero;

    const int lr = tid >> 2;        // 0..63 : tile row loaded by this thread
    const int lc = (tid & 3) * 8;   // 0,8,16,24 : k-offset (8 bf16 = 16 bytes)

    // ---- fragment build + 8 WMMAs from LDS buffer `buf` -------------------
    // A 16x32 bf16: lane half h, row lrow; elems 0..7 = K[h*8..], 8..15 = K[16+h*8..]
    // B 32x16 bf16: lane half h, col lrow; elems 0..7 = K[h*8..], 8..15 = K[16+h*8..]
    auto compute_tile = [&](int buf) {
        Frag fa[4];
#pragma unroll
        for (int tm = 0; tm < 4; ++tm) {
            const __bf16* base = &As[buf][(wm * 64 + tm * 16 + lrow) * BKP];
            fa[tm].u[0] = *(const uint4*)(base + half * 8);
            fa[tm].u[1] = *(const uint4*)(base + 16 + half * 8);
        }
        Frag fb[2];
#pragma unroll
        for (int tn = 0; tn < 2; ++tn) {
            const __bf16* base = &Ws[buf][(wn * 32 + tn * 16 + lrow) * BKP];
            fb[tn].u[0] = *(const uint4*)(base + half * 8);
            fb[tn].u[1] = *(const uint4*)(base + 16 + half * 8);
        }
#pragma unroll
        for (int tm = 0; tm < 4; ++tm)
#pragma unroll
            for (int tn = 0; tn < 2; ++tn)
                acc[tm][tn] = __builtin_amdgcn_wmma_f32_16x16x32_bf16(
                    false, fa[tm].v, false, fb[tn].v,
                    (short)0, acc[tm][tn], false, false);
    };

    if constexpr (!GUARD) {
        // ---- async double-buffered pipeline (M,N mult of 128, K mult of 32)
        const __bf16* aG0 = Ab + (long)(bm0 + lr)      * K + lc;
        const __bf16* aG1 = Ab + (long)(bm0 + lr + 64) * K + lc;
        const __bf16* wG0 = Wb + (long)(bn0 + lr)      * K + lc;
        const __bf16* wG1 = Wb + (long)(bn0 + lr + 64) * K + lc;

        auto issue_tile = [&](int k0, int buf) {
            // LDS dest = low 32 bits of the generic address (LDS aperture rule)
            unsigned la0 = (unsigned)(uintptr_t)(&As[buf][lr * BKP + lc]);
            unsigned la1 = (unsigned)(uintptr_t)(&As[buf][(lr + 64) * BKP + lc]);
            unsigned lw0 = (unsigned)(uintptr_t)(&Ws[buf][lr * BKP + lc]);
            unsigned lw1 = (unsigned)(uintptr_t)(&Ws[buf][(lr + 64) * BKP + lc]);
            asm volatile("global_load_async_to_lds_b128 %0, %1, off"
                         :: "v"(la0), "v"((unsigned long long)(uintptr_t)(aG0 + k0))
                         : "memory");
            asm volatile("global_load_async_to_lds_b128 %0, %1, off"
                         :: "v"(la1), "v"((unsigned long long)(uintptr_t)(aG1 + k0))
                         : "memory");
            asm volatile("global_load_async_to_lds_b128 %0, %1, off"
                         :: "v"(lw0), "v"((unsigned long long)(uintptr_t)(wG0 + k0))
                         : "memory");
            asm volatile("global_load_async_to_lds_b128 %0, %1, off"
                         :: "v"(lw1), "v"((unsigned long long)(uintptr_t)(wG1 + k0))
                         : "memory");
        };

        const int nk = K / BK;        // >= 1 (K multiple of 32)
        issue_tile(0, 0);
        // steady state: unconditional issue-ahead + wait for the older 4 loads
        for (int i = 0; i + 1 < nk; ++i) {
            const int cur = i & 1;
            issue_tile((i + 1) * BK, cur ^ 1);
            // tile i's 4 loads retire first (in-order); tile i+1's 4 stay in flight
            asm volatile("s_wait_asynccnt 0x4" ::: "memory");
            __syncthreads();            // all waves: tile i resident in LDS
            compute_tile(cur);
            __syncthreads();            // LDS buffer reusable for tile i+2
        }
        // drain: last tile
        asm volatile("s_wait_asynccnt 0x0" ::: "memory");
        __syncthreads();
        compute_tile((nk - 1) & 1);
    } else {
        // ---- guarded synchronous path (only the logits GEMM, N=45) --------
        for (int k0 = 0; k0 < K; k0 += BK) {
#pragma unroll
            for (int p = 0; p < 2; ++p) {
                const int r  = lr + p * 64;
                const int gm = bm0 + r;
                uint4 va = {0u, 0u, 0u, 0u};
                if (gm < M) va = *(const uint4*)(Ab + (long)gm * K + k0 + lc);
                *(uint4*)(&As[0][r * BKP + lc]) = va;

                const int gn = bn0 + r;
                uint4 vw = {0u, 0u, 0u, 0u};
                if (gn < N) vw = *(const uint4*)(Wb + (long)gn * K + k0 + lc);
                *(uint4*)(&Ws[0][r * BKP + lc]) = vw;
            }
            if (k0 + BK < K) {  // speculative prefetch -> global_prefetch
                __builtin_prefetch(Ab + (long)(bm0 + lr) * K + k0 + BK + lc, 0, 1);
                __builtin_prefetch(Wb + (long)(bn0 + lr) * K + k0 + BK + lc, 0, 1);
            }
            __syncthreads();
            compute_tile(0);
            __syncthreads();
        }
    }

    // ---- epilogue: C/D layout: VGPR v holds row (v + 8*half), col lrow ----
#pragma unroll
    for (int tm = 0; tm < 4; ++tm) {
#pragma unroll
        for (int tn = 0; tn < 2; ++tn) {
            const int gn = bn0 + wn * 32 + tn * 16 + lrow;
            if (gn >= N) continue;
            const int gmBase = bm0 + wm * 64 + tm * 16 + half * 8;
            const float bval = (bias != nullptr) ? bias[gn] : 0.f;
#pragma unroll
            for (int v = 0; v < 8; ++v) {
                const int gm = gmBase + v;
                if (gm >= M) continue;
                float a = acc[tm][tn][v];
                if constexpr (EPI == EPI_F32) {
                    outF[(long)gm * N + gn] = a + bval;
                } else if constexpr (EPI == EPI_GELU) {
                    float t = a + bval;
                    float g = 0.5f * t * (1.f + erff(t * 0.70710678118654752f));
                    outB[(long)gm * N + gn] = (__bf16)g;
                } else if constexpr (EPI == EPI_QKV) {
                    // gm = b*SEQ + s ; gn in [0,3E): q|k|v, head, dim
                    float t = a + bval;
                    const int b    = gm >> 8;
                    const int s    = gm & 255;
                    const int part = gn >> 10;        // 0=q 1=k 2=v
                    const int rem  = gn & 1023;
                    const int h    = rem >> 7;
                    const int d    = rem & 127;
                    const long bh  = (long)(b * NH + h);
                    if (part == 0)      outB [(bh * SEQ + s) * DH + d] = (__bf16)t;
                    else if (part == 1) outB2[(bh * SEQ + s) * DH + d] = (__bf16)t;
                    else                outB3[(bh * DH + d) * SEQ + s] = (__bf16)t; // V^T
                } else if constexpr (EPI == EPI_SCORES) {
                    // z = b*NH + h ; gm = query, gn = key
                    const int h = blockIdx.z & (NH - 1);
                    float t = a * scale;
                    if (gn <= gm) {
                        int dist = gm - gn;
                        if (dist > MAXD) dist = MAXD;
                        t += distEmb[dist * NH + h];
                    } else {
                        t = -1e9f;
                    }
                    outF[(long)blockIdx.z * SEQ * SEQ + (long)gm * SEQ + gn] = t;
                } else if constexpr (EPI == EPI_ATTNOUT) {
                    const int b = blockIdx.z >> 3;
                    const int h = blockIdx.z & (NH - 1);
                    outB[((long)(b * SEQ + gm)) * E + h * DH + gn] = (__bf16)a;
                }
            }
        }
    }
}

// ---------------------------------------------------------------------------
// fp32 -> bf16 conversion (n must be a multiple of 4; all our sizes are)
// ---------------------------------------------------------------------------
__global__ void cvt_kernel(const float* __restrict__ in, __bf16* __restrict__ out, long n)
{
    const long nv = n >> 2;
    for (long i = (long)blockIdx.x * blockDim.x + threadIdx.x; i < nv;
         i += (long)gridDim.x * blockDim.x) {
        float4 f = ((const float4*)in)[i];
        union { __bf16 h[4]; unsigned long long u; } pk;
        pk.h[0] = (__bf16)f.x; pk.h[1] = (__bf16)f.y;
        pk.h[2] = (__bf16)f.z; pk.h[3] = (__bf16)f.w;
        ((unsigned long long*)out)[i] = pk.u;
    }
}

// ---------------------------------------------------------------------------
// Embedding: x[m,:] = token_emb[seq[m]] * sqrt(E);  also bf16 copy
// ---------------------------------------------------------------------------
__global__ __launch_bounds__(256) void embed_kernel(
    const int* __restrict__ seq, const float* __restrict__ emb,
    float* __restrict__ x, __bf16* __restrict__ xb)
{
    const long m   = blockIdx.x;
    const int  tok = seq[m];
    const float* er = emb + (long)tok * E;
    for (int i = threadIdx.x; i < E; i += blockDim.x) {
        float v = er[i] * 32.0f;   // sqrt(1024)
        x [m * E + i] = v;
        xb[m * E + i] = (__bf16)v;
    }
}

// ---------------------------------------------------------------------------
// Softmax over rows of 256 (one wave32 per row, 8 elems per lane)
// ---------------------------------------------------------------------------
__global__ __launch_bounds__(256) void softmax_kernel(
    const float* __restrict__ sc, __bf16* __restrict__ pr)
{
    const long row  = (long)blockIdx.x * 8 + (threadIdx.x >> 5);
    const int  lane = threadIdx.x & 31;
    const float* r  = sc + row * SEQ + lane * 8;
    float v[8];
    float mx = -3.4e38f;
#pragma unroll
    for (int i = 0; i < 8; ++i) { v[i] = r[i]; mx = fmaxf(mx, v[i]); }
#pragma unroll
    for (int o = 16; o > 0; o >>= 1) mx = fmaxf(mx, __shfl_xor(mx, o, 32));
    float sum = 0.f;
#pragma unroll
    for (int i = 0; i < 8; ++i) { v[i] = expf(v[i] - mx); sum += v[i]; }
#pragma unroll
    for (int o = 16; o > 0; o >>= 1) sum += __shfl_xor(sum, o, 32);
    const float inv = 1.f / sum;
    __bf16* w = pr + row * SEQ + lane * 8;
#pragma unroll
    for (int i = 0; i < 8; ++i) w[i] = (__bf16)(v[i] * inv);
}

// ---------------------------------------------------------------------------
// Residual + LayerNorm over E=1024 (one 256-thread block per row)
// ---------------------------------------------------------------------------
__device__ __forceinline__ float block_sum(float v, float* sm)
{
#pragma unroll
    for (int o = 16; o > 0; o >>= 1) v += __shfl_xor(v, o, 32);
    const int w = threadIdx.x >> 5, lane = threadIdx.x & 31;
    if (lane == 0) sm[w] = v;
    __syncthreads();
    float r = 0.f;
    if (threadIdx.x < 8) r = sm[threadIdx.x];
    if (w == 0) {
#pragma unroll
        for (int o = 4; o > 0; o >>= 1) r += __shfl_xor(r, o, 32);
        if (lane == 0) sm[8] = r;
    }
    __syncthreads();
    r = sm[8];
    __syncthreads();
    return r;
}

__global__ __launch_bounds__(256) void add_ln_kernel(
    const float* __restrict__ x, const float* __restrict__ y,
    const float* __restrict__ g, const float* __restrict__ b,
    float* __restrict__ xo, __bf16* __restrict__ xob)
{
    __shared__ float sm[9];
    const long row = blockIdx.x;
    const float* xr = x + row * E;
    float v[4];
#pragma unroll
    for (int i = 0; i < 4; ++i) v[i] = xr[threadIdx.x + 256 * i];
    if (y != nullptr) {
        const float* yr = y + row * E;
#pragma unroll
        for (int i = 0; i < 4; ++i) v[i] += yr[threadIdx.x + 256 * i];
    }
    float s = v[0] + v[1] + v[2] + v[3];
    const float mean = block_sum(s, sm) * (1.f / (float)E);
    float d = 0.f;
#pragma unroll
    for (int i = 0; i < 4; ++i) { float t = v[i] - mean; d += t * t; }
    const float var = block_sum(d, sm) * (1.f / (float)E);
    const float inv = rsqrtf(var + 1e-5f);
#pragma unroll
    for (int i = 0; i < 4; ++i) {
        const int e = threadIdx.x + 256 * i;
        float o = (v[i] - mean) * inv * g[e] + b[e];
        xo [row * E + e] = o;
        xob[row * E + e] = (__bf16)o;
    }
}

// ---------------------------------------------------------------------------
// Host orchestration
// ---------------------------------------------------------------------------
extern "C" void kernel_launch(void* const* d_in, const int* in_sizes, int n_in,
                              void* d_out, int out_size, void* d_ws, size_t ws_size,
                              hipStream_t stream)
{
    (void)in_sizes; (void)n_in; (void)out_size; (void)ws_size;

    const int*   seq      = (const int*)  d_in[0];
    const float* tokEmb   = (const float*)d_in[1];
    const float* distEmb  = (const float*)d_in[2];
    const float* Wqkv     = (const float*)d_in[3];
    const float* bqkv     = (const float*)d_in[4];
    const float* Wo       = (const float*)d_in[5];
    const float* bo       = (const float*)d_in[6];
    const float* W1       = (const float*)d_in[7];
    const float* b1       = (const float*)d_in[8];
    const float* W2       = (const float*)d_in[9];
    const float* b2       = (const float*)d_in[10];
    const float* ln1_g    = (const float*)d_in[11];
    const float* ln1_b    = (const float*)d_in[12];
    const float* ln2_g    = (const float*)d_in[13];
    const float* ln2_b    = (const float*)d_in[14];
    const float* fn_g     = (const float*)d_in[15];
    const float* fn_b     = (const float*)d_in[16];
    const float* genW     = (const float*)d_in[17];
    const float* genB     = (const float*)d_in[18];
    float* out = (float*)d_out;

    // ---- workspace carve-up ----
    uintptr_t p = (uintptr_t)d_ws;
    auto take = [&](size_t bytes) -> void* {
        void* r = (void*)p;
        p += (bytes + 255) & ~(size_t)255;
        return r;
    };
    const size_t nWqkv = (size_t)NL * 3 * E * E;
    const size_t nWo   = (size_t)NL * E * E;
    const size_t nW1   = (size_t)NL * FF * E;
    const size_t nW2   = (size_t)NL * E * FF;
    const size_t nGenW = (size_t)VOC * E;

    __bf16* wqkvB = (__bf16*)take(nWqkv * 2);
    __bf16* woB   = (__bf16*)take(nWo   * 2);
    __bf16* w1B   = (__bf16*)take(nW1   * 2);
    __bf16* w2B   = (__bf16*)take(nW2   * 2);
    __bf16* genWB = (__bf16*)take(nGenW * 2);

    float*  x      = (float*) take((size_t)MTOK * E * 4);
    float*  y      = (float*) take((size_t)MTOK * E * 4);
    __bf16* xb     = (__bf16*)take((size_t)MTOK * E * 2);
    __bf16* qb     = (__bf16*)take((size_t)BSZ * NH * SEQ * DH * 2);
    __bf16* kb     = (__bf16*)take((size_t)BSZ * NH * SEQ * DH * 2);
    __bf16* vtb    = (__bf16*)take((size_t)BSZ * NH * DH * SEQ * 2);
    float*  scores = (float*) take((size_t)BSZ * NH * SEQ * SEQ * 4);
    __bf16* probs  = (__bf16*)take((size_t)BSZ * NH * SEQ * SEQ * 2);
    __bf16* xoB    = (__bf16*)take((size_t)MTOK * E * 2);
    __bf16* hB     = (__bf16*)take((size_t)MTOK * FF * 2);

    const dim3 blk(256);
    auto cvt = [&](const float* in, __bf16* o, long n) {
        long blocks = (n / 4 + 255) / 256;
        if (blocks > 4096) blocks = 4096;
        cvt_kernel<<<(int)blocks, 256, 0, stream>>>(in, o, n);
    };

    // one-time (per launch) weight conversion to bf16
    cvt(Wqkv, wqkvB, (long)nWqkv);
    cvt(Wo,   woB,   (long)nWo);
    cvt(W1,   w1B,   (long)nW1);
    cvt(W2,   w2B,   (long)nW2);
    cvt(genW, genWB, (long)nGenW);

    // embedding
    embed_kernel<<<MTOK, blk, 0, stream>>>(seq, tokEmb, x, xb);

    const float scl = 0.08838834764831845f;   // 1/sqrt(DH)

    for (int l = 0; l < NL; ++l) {
        // QKV projection -> scatter q, k, v^T
        gemm_kernel<EPI_QKV, false><<<dim3(24, 64, 1), blk, 0, stream>>>(
            xb, wqkvB + (long)l * 3 * E * E, bqkv + (long)l * 3 * E,
            nullptr, qb, kb, vtb, MTOK, 3 * E, E, 0, 0, 1.f, nullptr);

        // scores = Q K^T * scale + rel-pos bias + causal mask   [B*H batched]
        gemm_kernel<EPI_SCORES, false><<<dim3(2, 2, BSZ * NH), blk, 0, stream>>>(
            qb, kb, nullptr, scores, nullptr, nullptr, nullptr,
            SEQ, SEQ, DH, (long)SEQ * DH, (long)SEQ * DH, scl, distEmb);

        // softmax rows
        softmax_kernel<<<(BSZ * NH * SEQ) / 8, blk, 0, stream>>>(scores, probs);

        // O = P V  (W = V^T so loads are contiguous), scatter to [B*S, E]
        gemm_kernel<EPI_ATTNOUT, false><<<dim3(1, 2, BSZ * NH), blk, 0, stream>>>(
            probs, vtb, nullptr, nullptr, xoB, nullptr, nullptr,
            SEQ, DH, SEQ, (long)SEQ * SEQ, (long)DH * SEQ, 1.f, nullptr);

        // output projection
        gemm_kernel<EPI_F32, false><<<dim3(8, 64, 1), blk, 0, stream>>>(
            xoB, woB + (long)l * E * E, bo + (long)l * E,
            y, nullptr, nullptr, nullptr, MTOK, E, E, 0, 0, 1.f, nullptr);

        // x = LN(x + attn_out)
        add_ln_kernel<<<MTOK, blk, 0, stream>>>(
            x, y, ln1_g + (long)l * E, ln1_b + (long)l * E, x, xb);

        // FFN up + exact GELU
        gemm_kernel<EPI_GELU, false><<<dim3(32, 64, 1), blk, 0, stream>>>(
            xb, w1B + (long)l * FF * E, b1 + (long)l * FF,
            nullptr, hB, nullptr, nullptr, MTOK, FF, E, 0, 0, 1.f, nullptr);

        // FFN down
        gemm_kernel<EPI_F32, false><<<dim3(8, 64, 1), blk, 0, stream>>>(
            hB, w2B + (long)l * E * FF, b2 + (long)l * E,
            y, nullptr, nullptr, nullptr, MTOK, E, FF, 0, 0, 1.f, nullptr);

        // x = LN(x + ffn_out)
        add_ln_kernel<<<MTOK, blk, 0, stream>>>(
            x, y, ln2_g + (long)l * E, ln2_b + (long)l * E, x, xb);
    }

    // final LN
    add_ln_kernel<<<MTOK, blk, 0, stream>>>(x, nullptr, fn_g, fn_b, x, xb);

    // logits: [8192, 45]  (guarded path handles N=45)
    gemm_kernel<EPI_F32, true><<<dim3(1, 64, 1), blk, 0, stream>>>(
        xb, genWB, genB, out, nullptr, nullptr, nullptr,
        MTOK, VOC, E, 0, 0, 1.f, nullptr);
}
